// PlugLSTM_75196287418441
// MI455X (gfx1250) — matrix operand order repaired and checked by reading, not compiled, not run on hardware
//
#include <hip/hip_runtime.h>
#include <hip/hip_bf16.h>
#include <cstdint>
#include <cstddef>

// ---------------- problem constants ----------------
#define HID    1024
#define BATCH  256
#define TSEQ   128
#define VOCABN 256
#define G4     (4 * HID)
#define KCHUNK 32
#define NCHUNK (HID / KCHUNK)
#define BUFSZ  8192

typedef __attribute__((ext_vector_type(16))) __bf16 v16bf;
typedef __attribute__((ext_vector_type(8)))  __bf16 v8bf;
typedef __attribute__((ext_vector_type(8)))  float  v8f;

// ---------------- WMMA helpers ----------------
__device__ __forceinline__ v8f wmma_bf16(v16bf a, v16bf b, v8f c) {
  // D = A(16x32 bf16) * B(32x16 bf16) + C(16x16 f32)
  return __builtin_amdgcn_wmma_f32_16x16x32_bf16(
      /*neg_a=*/false, a, /*neg_b=*/false, b,
      /*c_mod=*/(short)0, c, /*reuse_a=*/false, /*reuse_b=*/false);
}

// A fragment: 16x32 bf16, contiguous k order, same as B:
//   lanes 0-15  : row M=lane,    K = 0..15
//   lanes 16-31 : row M=lane-16, K = 16..31
__device__ __forceinline__ v16bf load_a_frag(const __bf16* row0, long ld,
                                             int k, int lane) {
  int m  = lane & 15;
  int hs = (lane >> 4) & 1;
  const __bf16* p = row0 + (size_t)m * (size_t)ld + k + hs * 16;
  v8bf lo = *(const v8bf*)(p);        // global_load_b128
  v8bf hi = *(const v8bf*)(p + 8);    // global_load_b128
  return __builtin_shufflevector(lo, hi, 0,1,2,3,4,5,6,7,8,9,10,11,12,13,14,15);
}

// B fragment straight from global (decoder): 32x16 bf16 (K x N).
// Column n of B == row (colbase+n) of row-major W (we compute x @ W^T):
//   lanes 0-15  : col N=lane,    K = 0..15
//   lanes 16-31 : col N=lane-16, K = 16..31
__device__ __forceinline__ v16bf load_b_frag(const __bf16* col0, long ld,
                                             int k, int lane) {
  int n  = lane & 15;
  int ko = ((lane >> 4) & 1) << 4;
  const __bf16* p = col0 + (size_t)n * (size_t)ld + k + ko;
  return *(const v16bf*)(p);          // 2x global_load_b128
}

__device__ __forceinline__ float sigmoidf_fast(float x) {
  return 1.0f / (1.0f + __expf(-x));
}

// ---------------- CDNA5 async global->LDS copy (ASYNCcnt path) ----------------
// Per-lane: LDS[lds_off] = MEM[gaddr] (16 bytes); one instruction moves
// 32 lanes x 16B = 512B.
__device__ __forceinline__ void async_copy_b128(uint32_t lds_off, const void* gsrc) {
  asm volatile("global_load_async_to_lds_b128 %0, %1, off"
               :: "v"(lds_off), "v"(gsrc)
               : "memory");
}
__device__ __forceinline__ void wait_async_le4() {
  asm volatile("s_wait_asynccnt 0x4" ::: "memory");
}
__device__ __forceinline__ void wait_async_0() {
  asm volatile("s_wait_asynccnt 0x0" ::: "memory");
}
// Split workgroup barrier without __syncthreads' implicit full memory fence
// (producer side is ordered by s_wait_asynccnt; consumer side by DScnt that
// the compiler tracks on the actual ds_load uses).
__device__ __forceinline__ void wg_barrier() {
  asm volatile("s_barrier_signal -1\n\ts_barrier_wait -1" ::: "memory");
}

// ---------------- utility kernels ----------------
__global__ void f2bf_kernel(const float* __restrict__ src,
                            __bf16* __restrict__ dst, int n) {
  int i = blockIdx.x * blockDim.x + threadIdx.x;
  int stride = gridDim.x * blockDim.x;
  for (; i < n; i += stride) dst[i] = (__bf16)src[i];
}

__global__ void zero_f32_kernel(float* __restrict__ p, int n) {
  int i = blockIdx.x * blockDim.x + threadIdx.x;
  int stride = gridDim.x * blockDim.x;
  for (; i < n; i += stride) p[i] = 0.0f;
}

__global__ void zero_bf16_kernel(__bf16* __restrict__ p, int n) {
  int i = blockIdx.x * blockDim.x + threadIdx.x;
  int stride = gridDim.x * blockDim.x;
  for (; i < n; i += stride) p[i] = (__bf16)0.0f;
}

__global__ void addvec_kernel(const float* __restrict__ a,
                              const float* __restrict__ b,
                              float* __restrict__ o, int n) {
  int i = blockIdx.x * blockDim.x + threadIdx.x;
  int stride = gridDim.x * blockDim.x;
  for (; i < n; i += stride) o[i] = a[i] + b[i];
}

// inp[b][t][:] = bf16( emb[token(b,t)][:] + y[b]*yW[:] + yb[:] )
__global__ void embed_kernel(const int* __restrict__ x,
                             const float* __restrict__ y,
                             const float* __restrict__ emb,
                             const float* __restrict__ yW,
                             const float* __restrict__ yb,
                             __bf16* __restrict__ inp) {
  int bt = blockIdx.x;            // 0 .. B*T-1
  int b = bt / TSEQ, t = bt % TSEQ;
  int tok = (t == 0) ? VOCABN : x[b * TSEQ + (t - 1)];
  float ys = y[b];
  const float* erow = emb + (size_t)tok * HID;
  __bf16* orow = inp + (size_t)bt * HID;
  int h0 = threadIdx.x * 4;
  #pragma unroll
  for (int j = 0; j < 4; ++j) {
    int h = h0 + j;
    orow[h] = (__bf16)(erow[h] + ys * yW[h] + yb[h]);
  }
}

// ---------------- fused LSTM timestep ----------------
// Block = 4 waves = 128 batch rows x 16 hidden cols x all 4 gates; each wave
// owns 32 rows (2 M-subtiles), so every LDS B-fragment feeds 2 WMMAs and
// every A pair feeds 4. Weight K-chunks (8KB: 4 gates x {Wih,Whh} x 16 cols
// x 32 k) are staged into LDS with async global->LDS copies, double-buffered
// at static offsets, and consumed via ds_load_b128 in preloaded groups so
// WMMAs issue back-to-back.
__global__ __launch_bounds__(128) void lstm_step_kernel(
    const __bf16* __restrict__ xin,  long xld,   // layer input rows (stride elems)
    const __bf16* __restrict__ hprev, long hld,  // h_{t-1} rows (stride 0 => zeros)
    const __bf16* __restrict__ Wih,              // [4H,H] bf16 row-major
    const __bf16* __restrict__ Whh,              // [4H,H] bf16 row-major
    const float* __restrict__ bsum,              // b_ih + b_hh, [4H] f32
    float* __restrict__ cstate,                  // [B,H] f32, updated in place
    __bf16* __restrict__ hout, long hold) {      // h_t rows (stride elems)
  // LDS buffer layout: row r = (g*2+m)*16 + n holds W_m[g*H+colbase+n][k..k+31]
  // (64B); B-frag lanes 0-15 read bytes [0,32), lanes 16-31 read [32,64).
  __shared__ __align__(64) char smem[2 * BUFSZ];

  const int tid  = threadIdx.x;
  const int lane = tid & 31;
  const int wave = tid >> 5;
  const int colbase = (blockIdx.x & 63) << 4;                 // 64 col tiles
  const int rowbase = ((blockIdx.x >> 6) << 7) + (wave << 5); // 2 grp x (4 waves x 32 rows)

  const __bf16* xrow0 = xin   + (size_t)(rowbase +  0) * (size_t)xld;
  const __bf16* xrow1 = xin   + (size_t)(rowbase + 16) * (size_t)xld;
  const __bf16* hrow0 = hprev + (size_t)(rowbase +  0) * (size_t)hld;
  const __bf16* hrow1 = hprev + (size_t)(rowbase + 16) * (size_t)hld;

  // Per-lane staging descriptors: 4 async b128 per wave per K-chunk.
  const int cq = tid & 3;                    // 16B chunk within a 64B row
  uint32_t ldsoff[4];
  const __bf16* gsrc[4];
  #pragma unroll
  for (int j = 0; j < 4; ++j) {
    int r  = (tid >> 2) + j * 32;            // 0..127
    int gm = r >> 4;                         // gate*2 + matrix
    int n  = r & 15;
    int g  = gm >> 1, m = gm & 1;
    const __bf16* wbase = (m == 0) ? Wih : Whh;
    gsrc[j]   = wbase + (size_t)(g * HID + colbase + n) * HID + cq * 8;
    ldsoff[j] = (uint32_t)(r * 64 + cq * 16);
  }
  const uint32_t smembase = (uint32_t)(uintptr_t)(void*)&smem[0];

  v8f acc[2][4] = {{v8f{}, v8f{}, v8f{}, v8f{}},
                   {v8f{}, v8f{}, v8f{}, v8f{}}};

  auto stage = [&](uint32_t bufoff, int k) {
    #pragma unroll
    for (int j = 0; j < 4; ++j)
      async_copy_b128(smembase + bufoff + ldsoff[j], gsrc[j] + (size_t)k);
  };

  const int nlo = lane & 15;
  const int hsb = ((lane >> 4) & 1) * 32;
  auto compute = [&](uint32_t bufoff, int k) {
    v16bf ax0 = load_a_frag(xrow0, xld, k, lane);
    v16bf ax1 = load_a_frag(xrow1, xld, k, lane);
    v16bf ah0 = load_a_frag(hrow0, hld, k, lane);
    v16bf ah1 = load_a_frag(hrow1, hld, k, lane);
    const char* bufp = smem + bufoff;
    v16bf bf_[4];
    #pragma unroll
    for (int g = 0; g < 4; ++g)
      bf_[g] = *(const v16bf*)(bufp + ((g * 2 + 0) * 16 + nlo) * 64 + hsb);
    #pragma unroll
    for (int g = 0; g < 4; ++g) {
      acc[0][g] = wmma_bf16(ax0, bf_[g], acc[0][g]);
      acc[1][g] = wmma_bf16(ax1, bf_[g], acc[1][g]);
    }
    #pragma unroll
    for (int g = 0; g < 4; ++g)
      bf_[g] = *(const v16bf*)(bufp + ((g * 2 + 1) * 16 + nlo) * 64 + hsb);
    #pragma unroll
    for (int g = 0; g < 4; ++g) {
      acc[0][g] = wmma_bf16(ah0, bf_[g], acc[0][g]);
      acc[1][g] = wmma_bf16(ah1, bf_[g], acc[1][g]);
    }
  };

  stage(0, 0);
  // Manual 2x unroll so both LDS buffer offsets are compile-time constants
  // (avoids accumulator phi shuffles on a dynamic buffer index).
  #pragma unroll 1
  for (int kc = 0; kc < NCHUNK; kc += 2) {
    const int k0 = kc * KCHUNK;
    // even chunk lives in buffer 0
    if (kc + 1 < NCHUNK) { stage(BUFSZ, k0 + KCHUNK); wait_async_le4(); }
    else                 { wait_async_0(); }
    wg_barrier();
    compute(0, k0);
    wg_barrier();
    // odd chunk lives in buffer 1
    if (kc + 2 < NCHUNK) { stage(0, k0 + 2 * KCHUNK); wait_async_le4(); }
    else                 { wait_async_0(); }
    wg_barrier();
    compute(BUFSZ, k0 + KCHUNK);
    wg_barrier();
  }

  // Epilogue. C/D layout: VGPR e, lanes 0-15 -> M=e, lanes 16-31 -> M=e+8;
  // N = lane & 15.
  const int moff = ((lane >> 4) & 1) * 8;
  const int hcol = colbase + nlo;
  const float bi = bsum[0 * HID + hcol];
  const float bf = bsum[1 * HID + hcol];
  const float bg = bsum[2 * HID + hcol];
  const float bo = bsum[3 * HID + hcol];

  #pragma unroll
  for (int rt = 0; rt < 2; ++rt) {
    #pragma unroll
    for (int e = 0; e < 8; ++e) {
      const int brow = rowbase + rt * 16 + e + moff;
      const float ig = sigmoidf_fast(acc[rt][0][e] + bi);
      const float fg = sigmoidf_fast(acc[rt][1][e] + bf);
      const float gg = tanhf(acc[rt][2][e] + bg);
      const float og = sigmoidf_fast(acc[rt][3][e] + bo);
      const size_t ci = (size_t)brow * HID + hcol;
      const float cn = fg * cstate[ci] + ig * gg;
      cstate[ci] = cn;
      hout[(size_t)brow * (size_t)hold + hcol] = (__bf16)(og * tanhf(cn));
    }
  }
}

// ---------------- decoder GEMM ----------------
// logits[M=B*T, N=VOCAB] = h1[M,H] @ decW^T[H,V] + decb
__global__ __launch_bounds__(256) void decoder_kernel(
    const __bf16* __restrict__ h1,    // [B*T, H] bf16 row-major
    const __bf16* __restrict__ Wd,    // [V, H]   bf16 row-major
    const float* __restrict__ bd,
    float* __restrict__ out) {        // [B*T, V] f32
  const int lane = threadIdx.x & 31;
  const int wave = threadIdx.x >> 5;
  const int gid  = blockIdx.x * 8 + wave;   // 2048 x 16 = 32768 wave tiles
  const int rowbase = (gid >> 4) << 4;
  const int colbase = (gid & 15) << 4;

  const __bf16* arow = h1 + (size_t)rowbase * HID;
  const __bf16* bcol = Wd + (size_t)colbase * HID;

  v8f acc = {};
  for (int k = 0; k < HID; k += KCHUNK) {
    v16bf a = load_a_frag(arow, HID, k, lane);
    v16bf b = load_b_frag(bcol, HID, k, lane);
    acc = wmma_bf16(a, b, acc);
  }

  const int n    = lane & 15;
  const int moff = ((lane >> 4) & 1) * 8;
  const int vcol = colbase + n;
  const float bias = bd[vcol];
  #pragma unroll
  for (int e = 0; e < 8; ++e) {
    const int mrow = rowbase + e + moff;
    out[(size_t)mrow * VOCABN + vcol] = acc[e] + bias;
  }
}

// ---------------- host orchestration ----------------
extern "C" void kernel_launch(void* const* d_in, const int* in_sizes, int n_in,
                              void* d_out, int out_size, void* d_ws, size_t ws_size,
                              hipStream_t stream) {
  (void)in_sizes; (void)n_in; (void)out_size; (void)ws_size;
  const int*   x     = (const int*)  d_in[0];
  const float* y     = (const float*)d_in[1];
  const float* emb   = (const float*)d_in[2];
  const float* yW    = (const float*)d_in[3];
  const float* yb    = (const float*)d_in[4];
  const float* W_ih0 = (const float*)d_in[5];
  const float* W_hh0 = (const float*)d_in[6];
  const float* b_ih0 = (const float*)d_in[7];
  const float* b_hh0 = (const float*)d_in[8];
  const float* W_ih1 = (const float*)d_in[9];
  const float* W_hh1 = (const float*)d_in[10];
  const float* b_ih1 = (const float*)d_in[11];
  const float* b_hh1 = (const float*)d_in[12];
  const float* decW  = (const float*)d_in[13];
  const float* decb  = (const float*)d_in[14];
  float* out = (float*)d_out;

  // workspace carve-out (256B aligned)
  char* ws = (char*)d_ws;
  size_t off = 0;
  auto carve = [&](size_t bytes) -> char* {
    char* p = ws + off;
    off = (off + bytes + 255) & ~(size_t)255;
    return p;
  };
  __bf16* wb_ih0 = (__bf16*)carve((size_t)G4 * HID * 2);
  __bf16* wb_hh0 = (__bf16*)carve((size_t)G4 * HID * 2);
  __bf16* wb_ih1 = (__bf16*)carve((size_t)G4 * HID * 2);
  __bf16* wb_hh1 = (__bf16*)carve((size_t)G4 * HID * 2);
  __bf16* wb_dec = (__bf16*)carve((size_t)VOCABN * HID * 2);
  __bf16* inp    = (__bf16*)carve((size_t)BATCH * TSEQ * HID * 2);
  __bf16* h0_all = (__bf16*)carve((size_t)BATCH * TSEQ * HID * 2);
  __bf16* h1_all = (__bf16*)carve((size_t)BATCH * TSEQ * HID * 2);
  float*  c0     = (float*) carve((size_t)BATCH * HID * 4);
  float*  c1     = (float*) carve((size_t)BATCH * HID * 4);
  float*  bsum0  = (float*) carve((size_t)G4 * 4);
  float*  bsum1  = (float*) carve((size_t)G4 * 4);
  __bf16* zrow   = (__bf16*)carve((size_t)HID * 2);

  // 1) weights -> bf16; biases pre-combined
  f2bf_kernel<<<1024, 256, 0, stream>>>(W_ih0, wb_ih0, G4 * HID);
  f2bf_kernel<<<1024, 256, 0, stream>>>(W_hh0, wb_hh0, G4 * HID);
  f2bf_kernel<<<1024, 256, 0, stream>>>(W_ih1, wb_ih1, G4 * HID);
  f2bf_kernel<<<1024, 256, 0, stream>>>(W_hh1, wb_hh1, G4 * HID);
  f2bf_kernel<<<256,  256, 0, stream>>>(decW,  wb_dec, VOCABN * HID);
  addvec_kernel<<<16, 256, 0, stream>>>(b_ih0, b_hh0, bsum0, G4);
  addvec_kernel<<<16, 256, 0, stream>>>(b_ih1, b_hh1, bsum1, G4);

  // 2) init states
  zero_f32_kernel <<<256, 256, 0, stream>>>(c0, BATCH * HID);
  zero_f32_kernel <<<256, 256, 0, stream>>>(c1, BATCH * HID);
  zero_bf16_kernel<<<1,   256, 0, stream>>>(zrow, HID);

  // 3) embedding + y conditioning -> bf16 activations
  embed_kernel<<<BATCH * TSEQ, 256, 0, stream>>>(x, y, emb, yW, yb, inp);

  const long seq_ld = (long)TSEQ * HID;   // row stride within [B,T,H]

  // 4) layer 0 recurrence
  for (int t = 0; t < TSEQ; ++t) {
    const __bf16* hp  = (t == 0) ? zrow : (h0_all + (size_t)(t - 1) * HID);
    const long    hld = (t == 0) ? 0    : seq_ld;
    lstm_step_kernel<<<128, 128, 0, stream>>>(
        inp + (size_t)t * HID, seq_ld, hp, hld,
        wb_ih0, wb_hh0, bsum0, c0,
        h0_all + (size_t)t * HID, seq_ld);
  }

  // 5) layer 1 recurrence (input = layer-0 hidden states)
  for (int t = 0; t < TSEQ; ++t) {
    const __bf16* hp  = (t == 0) ? zrow : (h1_all + (size_t)(t - 1) * HID);
    const long    hld = (t == 0) ? 0    : seq_ld;
    lstm_step_kernel<<<128, 128, 0, stream>>>(
        h0_all + (size_t)t * HID, seq_ld, hp, hld,
        wb_ih1, wb_hh1, bsum1, c1,
        h1_all + (size_t)t * HID, seq_ld);
  }

  // 6) decoder: [32768,1024] @ [1024,256]^T + bias -> logits
  decoder_kernel<<<4096, 256, 0, stream>>>(h1_all, wb_dec, decb, out);
}
